// CrossAttentionMeshGraphNet_32169305047412
// MI455X (gfx1250) — matrix operand order repaired and checked by reading, not compiled, not run on hardware
//
#include <hip/hip_runtime.h>

// ---------------------------------------------------------------------------
// CDNA5 (gfx1250, wave32) MeshGraphNet forward.
// All H=128-wide matmuls via v_wmma_f32_16x16x32_f16 (f16 in, f32 accum).
// Each wave owns a 16-row tile; 8 n-tiles of v8f accumulators.
// B fragments for a whole k-step are preloaded into 8 distinct register
// octets so one s_wait covers an 8-WMMA burst (no per-WMMA load stall).
// Weights pre-packed into WMMA B-fragment layout; hidden tile transposed
// C-layout -> A-layout through a private LDS slice.
// Segment-sum fused into the edge kernel via global_atomic_add_f32.
// ---------------------------------------------------------------------------

typedef __attribute__((ext_vector_type(16))) _Float16 v16h;
typedef __attribute__((ext_vector_type(8)))  _Float16 v8h;
typedef __attribute__((ext_vector_type(8)))  float    v8f;
typedef __attribute__((ext_vector_type(4)))  float    v4f;

__device__ __forceinline__ v8f wmma_f16(v16h a, v16h b, v8f c) {
  // D = A(16x32 f16) * B(32x16 f16) + C(16x16 f32)
  return __builtin_amdgcn_wmma_f32_16x16x32_f16(
      /*neg_a=*/false, a, /*neg_b=*/false, b,
      /*c_mod=*/(short)0, c, /*reuse_a=*/false, /*reuse_b=*/false);
}

// A-fragment from a f32 row (32 valid cols starting at cb).
// Lane layout: elements 0..7 <- cols [cb+8*khalf, +8), 8..15 <- [cb+8*khalf+16, +8)
__device__ __forceinline__ v16h a_frag_f32(const float* p, int cb, int khalf) {
  const v4f* q0 = (const v4f*)(p + cb + 8 * khalf);
  const v4f* q1 = (const v4f*)(p + cb + 8 * khalf + 16);
  v4f r0 = q0[0], r1 = q0[1], r2 = q1[0], r3 = q1[1];
  v16h a;
#pragma unroll
  for (int i = 0; i < 4; ++i) {
    a[i]      = (_Float16)r0[i];
    a[4 + i]  = (_Float16)r1[i];
    a[8 + i]  = (_Float16)r2[i];
    a[12 + i] = (_Float16)r3[i];
  }
  return a;
}

// A-fragment from a f16 row (global or LDS).
__device__ __forceinline__ v16h a_frag_f16(const _Float16* p, int cb, int khalf) {
  v8h r0 = *(const v8h*)(p + cb + 8 * khalf);
  v8h r1 = *(const v8h*)(p + cb + 8 * khalf + 16);
  v16h a;
#pragma unroll
  for (int i = 0; i < 8; ++i) { a[i] = r0[i]; a[8 + i] = r1[i]; }
  return a;
}

// Guarded A-fragment for small first-layer K (pad to 32 with zeros).
__device__ __forceinline__ v16h a_frag_guard(const float* p, int khalf, int kin) {
  v16h a;
#pragma unroll
  for (int e = 0; e < 16; ++e) {
    int k = 8 * khalf + (e < 8 ? e : e + 8);
    a[e] = (k < kin) ? (_Float16)p[k] : (_Float16)0.0f;
  }
  return a;
}

// 8-WMMA burst for one k-step: preload all 8 B fragments into distinct
// registers (one clause, one wait), then issue 8 back-to-back WMMAs.
__device__ __forceinline__ void kstep8(v16h a, const v16h* __restrict__ wp,
                                       int ks, int lane, v8f acc[8]) {
  v16h b[8];
#pragma unroll
  for (int n = 0; n < 8; ++n) b[n] = wp[(ks * 8 + n) * 32 + lane];
#pragma unroll
  for (int n = 0; n < 8; ++n) acc[n] = wmma_f16(a, b[n], acc[n]);
}

// ---------------------------------------------------------------------------
// Pack f32 weight matrix W[K][128] into WMMA B-fragment layout (f16).
// Fragment (ks, nt): lane l -> col = nt*16 + (l&15), khalf = l>>4,
// element e -> K = ks*32 + 8*khalf + (e<8 ? e : e+8). One wave per fragment.
// ---------------------------------------------------------------------------
__global__ void pack_w_kernel(const float* __restrict__ w, int kin,
                              v16h* __restrict__ outp) {
  int frag = blockIdx.x;
  int ks = frag >> 3, nt = frag & 7;
  int l = threadIdx.x;
  int khalf = l >> 4;
  int col = nt * 16 + (l & 15);
  v16h v;
#pragma unroll
  for (int e = 0; e < 16; ++e) {
    int k = ks * 32 + khalf * 8 + (e < 8 ? e : e + 8);
    v[e] = (k < kin) ? (_Float16)w[k * 128 + col] : (_Float16)0.0f;
  }
  outp[frag * 32 + l] = v;
}

__global__ void zero_kernel(float* __restrict__ p, long n4) {
  long i = (long)blockIdx.x * 256 + threadIdx.x;  // one float4 per thread
  if (i < n4) ((v4f*)p)[i] = (v4f){0.f, 0.f, 0.f, 0.f};
}

// ---------------------------------------------------------------------------
// Tiny path: cond encoder + single-key attention (B=4). out_proj(v_proj(u)).
// One block of 512 threads; pure VALU, negligible cost.
// ---------------------------------------------------------------------------
__global__ void cond_attn_kernel(const float* __restrict__ cond,
                                 const float* __restrict__ w1, const float* __restrict__ b1,
                                 const float* __restrict__ w2, const float* __restrict__ b2,
                                 const float* __restrict__ wv, const float* __restrict__ bv,
                                 const float* __restrict__ wo, const float* __restrict__ bo,
                                 float* __restrict__ u_out, float* __restrict__ xc_out) {
  __shared__ float t0[4][128];
  __shared__ float t1[4][128];
  int t = threadIdx.x;
  int b = t >> 7, col = t & 127;
  float s = b1[col];
  for (int k = 0; k < 10; ++k) s += cond[b * 10 + k] * w1[k * 128 + col];
  t0[b][col] = fmaxf(s, 0.0f);
  __syncthreads();
  s = b2[col];
  for (int k = 0; k < 128; ++k) s += t0[b][k] * w2[k * 128 + col];
  u_out[b * 128 + col] = s;
  t1[b][col] = s;
  __syncthreads();
  s = bv[col];
  for (int k = 0; k < 128; ++k) s += t1[b][k] * wv[k * 128 + col];
  t0[b][col] = s;
  __syncthreads();
  s = bo[col];
  for (int k = 0; k < 128; ++k) s += t0[b][k] * wo[k * 128 + col];
  xc_out[b * 128 + col] = s;
}

// ---------------------------------------------------------------------------
// Encoder: out[M,128] = ReLU(in[M,kin] @ W1 + b1) @ W2 + b2.
// kin <= 32 (one guarded k-step), W2 is 128x128 (4 k-steps via LDS bounce).
// ---------------------------------------------------------------------------
template <typename OutT>
__global__ __launch_bounds__(256)
void encoder_mlp_kernel(const float* __restrict__ in, int kin, int M,
                        const v16h* __restrict__ w1p, const float* __restrict__ b1,
                        const v16h* __restrict__ w2p, const float* __restrict__ b2,
                        OutT* __restrict__ out) {
  __shared__ _Float16 hid[8][16][128];
  const int lane = threadIdx.x & 31;
  const int wave = threadIdx.x >> 5;
  const int lanelo = lane & 15, khalf = lane >> 4;
  const int tile = blockIdx.x * 8 + wave;
  if (tile * 16 >= M) return;
  const int row0 = tile * 16 + lanelo;

  v8f acc[8];
#pragma unroll
  for (int n = 0; n < 8; ++n) {
    float bb = b1[n * 16 + lanelo];
#pragma unroll
    for (int r = 0; r < 8; ++r) acc[n][r] = bb;
  }

  {  // pass 1 (K padded to 32)
    v16h a = a_frag_guard(in + (long)row0 * kin, khalf, kin);
    kstep8(a, w1p, 0, lane, acc);
  }
#pragma unroll
  for (int n = 0; n < 8; ++n) {
#pragma unroll
    for (int r = 0; r < 8; ++r)
      hid[wave][khalf * 8 + r][n * 16 + lanelo] = (_Float16)fmaxf(acc[n][r], 0.0f);
  }
#pragma unroll
  for (int n = 0; n < 8; ++n) {
    float bb = b2[n * 16 + lanelo];
#pragma unroll
    for (int r = 0; r < 8; ++r) acc[n][r] = bb;
  }
#pragma unroll
  for (int ks = 0; ks < 4; ++ks) {  // pass 2, K=128
    v16h a = a_frag_f16(&hid[wave][lanelo][0], ks * 32, khalf);
    kstep8(a, w2p, ks, lane, acc);
  }
#pragma unroll
  for (int n = 0; n < 8; ++n) {
#pragma unroll
    for (int r = 0; r < 8; ++r) {
      int m = khalf * 8 + r;
      out[(long)(tile * 16 + m) * 128 + n * 16 + lanelo] = (OutT)acc[n][r];
    }
  }
}

// ---------------------------------------------------------------------------
// Edge MLP: e' = MLP512(cat[h[row], h[col], e, u[batch[row]]]) ; fused
// scatter-add of e' into agg[row] via f32 atomics. e updated in place (f16).
// ---------------------------------------------------------------------------
__global__ __launch_bounds__(256)
void edge_mlp_kernel(const float* __restrict__ h, _Float16* e,
                     const int* __restrict__ erow, const int* __restrict__ ecol,
                     const int* __restrict__ batch, const float* __restrict__ u,
                     const v16h* __restrict__ w1p, const float* __restrict__ b1,
                     const v16h* __restrict__ w2p, const float* __restrict__ b2,
                     float* __restrict__ agg, int E) {
  __shared__ _Float16 hid[8][16][128];
  const int lane = threadIdx.x & 31;
  const int wave = threadIdx.x >> 5;
  const int lanelo = lane & 15, khalf = lane >> 4;
  const int tile = blockIdx.x * 8 + wave;
  if (tile * 16 >= E) return;
  const int ebase = tile * 16;
  const int eid0 = ebase + lanelo;
  const int r0 = erow[eid0], c0 = ecol[eid0];
  const int b0 = batch[r0];
  const float*    pa = h + (long)r0 * 128;
  const float*    pb = h + (long)c0 * 128;
  const _Float16* pe = e + (long)eid0 * 128;
  const float*    pu = u + b0 * 128;

  v8f acc[8];
#pragma unroll
  for (int n = 0; n < 8; ++n) {
    float bb = b1[n * 16 + lanelo];
#pragma unroll
    for (int r = 0; r < 8; ++r) acc[n][r] = bb;
  }
#pragma unroll
  for (int ks = 0; ks < 16; ++ks) {  // K = 512 = 4 chunks of 128
    const int chunk = ks >> 2;
    const int cb = (ks & 3) * 32;
    v16h a;
    if (chunk == 0)      a = a_frag_f32(pa, cb, khalf);
    else if (chunk == 1) a = a_frag_f32(pb, cb, khalf);
    else if (chunk == 2) a = a_frag_f16(pe, cb, khalf);
    else                 a = a_frag_f32(pu, cb, khalf);
    kstep8(a, w1p, ks, lane, acc);
  }
#pragma unroll
  for (int n = 0; n < 8; ++n) {
#pragma unroll
    for (int r = 0; r < 8; ++r)
      hid[wave][khalf * 8 + r][n * 16 + lanelo] = (_Float16)fmaxf(acc[n][r], 0.0f);
  }
#pragma unroll
  for (int n = 0; n < 8; ++n) {
    float bb = b2[n * 16 + lanelo];
#pragma unroll
    for (int r = 0; r < 8; ++r) acc[n][r] = bb;
  }
#pragma unroll
  for (int ks = 0; ks < 4; ++ks) {
    v16h a = a_frag_f16(&hid[wave][lanelo][0], ks * 32, khalf);
    kstep8(a, w2p, ks, lane, acc);
  }
  int er[8];
#pragma unroll
  for (int r = 0; r < 8; ++r) er[r] = erow[ebase + khalf * 8 + r];
#pragma unroll
  for (int n = 0; n < 8; ++n) {
    const int col = n * 16 + lanelo;
#pragma unroll
    for (int r = 0; r < 8; ++r) {
      float v = acc[n][r];
      int eid = ebase + khalf * 8 + r;
      e[(long)eid * 128 + col] = (_Float16)v;
      atomicAdd(&agg[(long)er[r] * 128 + col], v);  // segment_sum
    }
  }
}

// ---------------------------------------------------------------------------
// Node MLP: h += MLP384(cat[h, agg, x_cond[batch]]). In-place residual.
// ---------------------------------------------------------------------------
__global__ __launch_bounds__(256)
void node_mlp_kernel(float* h, const float* __restrict__ agg,
                     const float* __restrict__ xc, const int* __restrict__ batch,
                     const v16h* __restrict__ w1p, const float* __restrict__ b1,
                     const v16h* __restrict__ w2p, const float* __restrict__ b2,
                     int N) {
  __shared__ _Float16 hid[8][16][128];
  const int lane = threadIdx.x & 31;
  const int wave = threadIdx.x >> 5;
  const int lanelo = lane & 15, khalf = lane >> 4;
  const int tile = blockIdx.x * 8 + wave;
  if (tile * 16 >= N) return;
  const int node0 = tile * 16 + lanelo;
  const float* pa = h + (long)node0 * 128;
  const float* pg = agg + (long)node0 * 128;
  const float* pc = xc + (long)batch[node0] * 128;

  v8f acc[8];
#pragma unroll
  for (int n = 0; n < 8; ++n) {
    float bb = b1[n * 16 + lanelo];
#pragma unroll
    for (int r = 0; r < 8; ++r) acc[n][r] = bb;
  }
#pragma unroll
  for (int ks = 0; ks < 12; ++ks) {  // K = 384 = 3 chunks of 128
    const int chunk = ks >> 2;
    const int cb = (ks & 3) * 32;
    v16h a;
    if (chunk == 0)      a = a_frag_f32(pa, cb, khalf);
    else if (chunk == 1) a = a_frag_f32(pg, cb, khalf);
    else                 a = a_frag_f32(pc, cb, khalf);
    kstep8(a, w1p, ks, lane, acc);
  }
#pragma unroll
  for (int n = 0; n < 8; ++n) {
#pragma unroll
    for (int r = 0; r < 8; ++r)
      hid[wave][khalf * 8 + r][n * 16 + lanelo] = (_Float16)fmaxf(acc[n][r], 0.0f);
  }
#pragma unroll
  for (int n = 0; n < 8; ++n) {
    float bb = b2[n * 16 + lanelo];
#pragma unroll
    for (int r = 0; r < 8; ++r) acc[n][r] = bb;
  }
#pragma unroll
  for (int ks = 0; ks < 4; ++ks) {
    v16h a = a_frag_f16(&hid[wave][lanelo][0], ks * 32, khalf);
    kstep8(a, w2p, ks, lane, acc);
  }
#pragma unroll
  for (int n = 0; n < 8; ++n) {
    const int col = n * 16 + lanelo;
#pragma unroll
    for (int r = 0; r < 8; ++r) {
      int node = tile * 16 + khalf * 8 + r;
      float* dst = &h[(long)node * 128 + col];
      *dst = *dst + acc[n][r];  // residual
    }
  }
}

// ---------------------------------------------------------------------------
// Decoder: hidden = ReLU(h @ W1 + b1) via WMMA; 3-wide tail via VALU dots.
// ---------------------------------------------------------------------------
__global__ __launch_bounds__(256)
void decoder_kernel(const float* __restrict__ h,
                    const v16h* __restrict__ w1p, const float* __restrict__ b1,
                    const float* __restrict__ w2, const float* __restrict__ b2,
                    float* __restrict__ out, int N) {
  __shared__ _Float16 hid[8][16][128];
  const int lane = threadIdx.x & 31;
  const int wave = threadIdx.x >> 5;
  const int lanelo = lane & 15, khalf = lane >> 4;
  const int tile = blockIdx.x * 8 + wave;
  if (tile * 16 >= N) return;
  const float* pa = h + (long)(tile * 16 + lanelo) * 128;

  v8f acc[8];
#pragma unroll
  for (int n = 0; n < 8; ++n) {
    float bb = b1[n * 16 + lanelo];
#pragma unroll
    for (int r = 0; r < 8; ++r) acc[n][r] = bb;
  }
#pragma unroll
  for (int ks = 0; ks < 4; ++ks) {
    v16h a = a_frag_f32(pa, ks * 32, khalf);
    kstep8(a, w1p, ks, lane, acc);
  }
#pragma unroll
  for (int n = 0; n < 8; ++n) {
#pragma unroll
    for (int r = 0; r < 8; ++r)
      hid[wave][khalf * 8 + r][n * 16 + lanelo] = (_Float16)fmaxf(acc[n][r], 0.0f);
  }
  if (lane < 16) {
    const int m = lane;
    const int node = tile * 16 + m;
    float s0 = b2[0], s1 = b2[1], s2 = b2[2];
    for (int k = 0; k < 128; ++k) {
      float hv = (float)hid[wave][m][k];
      s0 += hv * w2[k * 3 + 0];
      s1 += hv * w2[k * 3 + 1];
      s2 += hv * w2[k * 3 + 2];
    }
    out[(long)node * 3 + 0] = s0;
    out[(long)node * 3 + 1] = s1;
    out[(long)node * 3 + 2] = s2;
  }
}

// ---------------------------------------------------------------------------
extern "C" void kernel_launch(void* const* d_in, const int* in_sizes, int n_in,
                              void* d_out, int out_size, void* d_ws, size_t ws_size,
                              hipStream_t stream) {
  const float* x     = (const float*)d_in[0];
  const int*   eidx  = (const int*)d_in[1];
  const float* eattr = (const float*)d_in[2];
  const float* cond  = (const float*)d_in[3];
  const int*   batch = (const int*)d_in[4];
  const float* enw1 = (const float*)d_in[5];
  const float* enb1 = (const float*)d_in[6];
  const float* enw2 = (const float*)d_in[7];
  const float* enb2 = (const float*)d_in[8];
  const float* eew1 = (const float*)d_in[9];
  const float* eeb1 = (const float*)d_in[10];
  const float* eew2 = (const float*)d_in[11];
  const float* eeb2 = (const float*)d_in[12];
  const float* ecw1 = (const float*)d_in[13];
  const float* ecb1 = (const float*)d_in[14];
  const float* ecw2 = (const float*)d_in[15];
  const float* ecb2 = (const float*)d_in[16];
  const float* wv   = (const float*)d_in[17];
  const float* bv   = (const float*)d_in[18];
  const float* wo   = (const float*)d_in[19];
  const float* bo   = (const float*)d_in[20];
  const float* pe_w1 = (const float*)d_in[21];
  const float* pe_b1 = (const float*)d_in[22];
  const float* pe_w2 = (const float*)d_in[23];
  const float* pe_b2 = (const float*)d_in[24];
  const float* pn_w1 = (const float*)d_in[25];
  const float* pn_b1 = (const float*)d_in[26];
  const float* pn_w2 = (const float*)d_in[27];
  const float* pn_b2 = (const float*)d_in[28];
  const float* dw1 = (const float*)d_in[29];
  const float* db1 = (const float*)d_in[30];
  const float* dw2 = (const float*)d_in[31];
  const float* db2 = (const float*)d_in[32];

  const int N = in_sizes[4];
  const int E = in_sizes[1] / 2;
  const int L = 4;

  char* ws = (char*)d_ws;
  size_t off = 0;
  auto alloc = [&](size_t bytes) -> void* {
    void* p = ws + off;
    off += (bytes + 255) & ~(size_t)255;
    return p;
  };
  float*    h    = (float*)alloc((size_t)N * 128 * sizeof(float));
  _Float16* ebuf = (_Float16*)alloc((size_t)E * 128 * sizeof(_Float16));
  float*    agg  = (float*)alloc((size_t)N * 128 * sizeof(float));
  float*    u    = (float*)alloc(512 * sizeof(float));
  float*    xc   = (float*)alloc(512 * sizeof(float));
  v16h* p_enw1 = (v16h*)alloc((size_t)8 * 32 * sizeof(v16h));
  v16h* p_enw2 = (v16h*)alloc((size_t)32 * 32 * sizeof(v16h));
  v16h* p_eew1 = (v16h*)alloc((size_t)8 * 32 * sizeof(v16h));
  v16h* p_eew2 = (v16h*)alloc((size_t)32 * 32 * sizeof(v16h));
  v16h* p_pe1[4]; v16h* p_pe2[4]; v16h* p_pn1[4]; v16h* p_pn2[4];
  for (int l = 0; l < 4; ++l) {
    p_pe1[l] = (v16h*)alloc((size_t)128 * 32 * sizeof(v16h));  // K=512: 16 ksteps
    p_pe2[l] = (v16h*)alloc((size_t)32 * 32 * sizeof(v16h));   // K=128
    p_pn1[l] = (v16h*)alloc((size_t)96 * 32 * sizeof(v16h));   // K=384: 12 ksteps
    p_pn2[l] = (v16h*)alloc((size_t)32 * 32 * sizeof(v16h));
  }
  v16h* p_dw1 = (v16h*)alloc((size_t)32 * 32 * sizeof(v16h));

  // --- pack weights into WMMA B-fragment layout (runs every call) ---
  pack_w_kernel<<<8, 32, 0, stream>>>(enw1, 12, p_enw1);
  pack_w_kernel<<<32, 32, 0, stream>>>(enw2, 128, p_enw2);
  pack_w_kernel<<<8, 32, 0, stream>>>(eew1, 4, p_eew1);
  pack_w_kernel<<<32, 32, 0, stream>>>(eew2, 128, p_eew2);
  for (int l = 0; l < 4; ++l) {
    pack_w_kernel<<<128, 32, 0, stream>>>(pe_w1 + (size_t)l * 512 * 128, 512, p_pe1[l]);
    pack_w_kernel<<<32, 32, 0, stream>>>(pe_w2 + (size_t)l * 128 * 128, 128, p_pe2[l]);
    pack_w_kernel<<<96, 32, 0, stream>>>(pn_w1 + (size_t)l * 384 * 128, 384, p_pn1[l]);
    pack_w_kernel<<<32, 32, 0, stream>>>(pn_w2 + (size_t)l * 128 * 128, 128, p_pn2[l]);
  }
  pack_w_kernel<<<32, 32, 0, stream>>>(dw1, 128, p_dw1);

  // --- tiny cond encoder + attention (B=4) ---
  cond_attn_kernel<<<1, 512, 0, stream>>>(cond, ecw1, ecb1, ecw2, ecb2,
                                          wv, bv, wo, bo, u, xc);

  const int ntiles = (N + 15) / 16, nblocks = (ntiles + 7) / 8;
  const int etiles = (E + 15) / 16, eblocks = (etiles + 7) / 8;

  // --- encoders ---
  encoder_mlp_kernel<float><<<nblocks, 256, 0, stream>>>(x, 12, N, p_enw1, enb1,
                                                         p_enw2, enb2, h);
  encoder_mlp_kernel<_Float16><<<eblocks, 256, 0, stream>>>(eattr, 4, E, p_eew1, eeb1,
                                                            p_eew2, eeb2, ebuf);

  // --- message-passing layers ---
  const long agg4 = (long)N * 128 / 4;  // float4 elements
  for (int l = 0; l < L; ++l) {
    zero_kernel<<<(int)((agg4 + 255) / 256), 256, 0, stream>>>(agg, agg4);
    edge_mlp_kernel<<<eblocks, 256, 0, stream>>>(h, ebuf, eidx, eidx + E, batch, u,
                                                 p_pe1[l], pe_b1 + l * 128,
                                                 p_pe2[l], pe_b2 + l * 128, agg, E);
    node_mlp_kernel<<<nblocks, 256, 0, stream>>>(h, agg, xc, batch,
                                                 p_pn1[l], pn_b1 + l * 128,
                                                 p_pn2[l], pn_b2 + l * 128, N);
  }

  // --- decoder ---
  decoder_kernel<<<nblocks, 256, 0, stream>>>(h, p_dw1, db1, dw2, db2,
                                              (float*)d_out, N);
}
